// NN_56934086476542
// MI455X (gfx1250) — matrix operand order repaired and checked by reading, not compiled, not run on hardware
//
#include <hip/hip_runtime.h>

typedef _Float16 h4   __attribute__((ext_vector_type(4)));
typedef _Float16 h8   __attribute__((ext_vector_type(8)));
typedef _Float16 v16h __attribute__((ext_vector_type(16)));
typedef float    v8f  __attribute__((ext_vector_type(8)));

#define BATCH   500000
#define IN_DIM  64
#define H_DIM   256
#define OUT_DIM 8
#define M_TILE  64
#define N_MT    4                       // 4 x 16-row sub-tiles
#define XS_STRIDE 72                    // 64 + 8 halves pad
#define HS_STRIDE 264                   // 256 + 8 halves pad
#define NTILES  ((BATCH + M_TILE - 1) / M_TILE)   // 7813, last tile has 32 rows

__device__ __forceinline__ v8f wmma_f16(v16h a, v16h b, v8f c) {
  return __builtin_amdgcn_wmma_f32_16x16x32_f16(
      false, a, false, b, (short)0, c, false, false);
}

// 16x32 f16 A-fragment from row-major f16 tile in LDS.
// lane<16 -> M=lane,   halves = K k0..k0+15, k0 = 32*ks
// lane>=16 -> M=lane-16, same with k0 = 32*ks + 16
__device__ __forceinline__ v16h make_afrag(const _Float16* base, int stride,
                                           int lane, int kstep) {
  const int M  = lane & 15;
  const int k0 = kstep * 32 + ((lane & 16) ? 16 : 0);
  const _Float16* p = base + M * stride + k0;
  h8 lo = *(const h8*)(p);
  h8 hi = *(const h8*)(p + 8);
  return __builtin_shufflevector(lo, hi, 0,1,2,3,4,5,6,7,8,9,10,11,12,13,14,15);
}

// Load one 32x16 f16 B-fragment: per lane 16 *contiguous* K values.
// lane<16 -> N=lane, K = kbase..kbase+15 ; lane>=16 -> N=lane-16, K += 16
__device__ __forceinline__ v16h load_bfrag(const float* __restrict__ wrow,
                                           int kbase) {
  const float4* p = (const float4*)(wrow + kbase);
  float4 q0 = p[0], q1 = p[1], q2 = p[2], q3 = p[3];
  v16h f;
  f[0]=(_Float16)q0.x; f[1]=(_Float16)q0.y; f[2]=(_Float16)q0.z; f[3]=(_Float16)q0.w;
  f[4]=(_Float16)q1.x; f[5]=(_Float16)q1.y; f[6]=(_Float16)q1.z; f[7]=(_Float16)q1.w;
  f[8]=(_Float16)q2.x; f[9]=(_Float16)q2.y; f[10]=(_Float16)q2.z; f[11]=(_Float16)q2.w;
  f[12]=(_Float16)q3.x; f[13]=(_Float16)q3.y; f[14]=(_Float16)q3.z; f[15]=(_Float16)q3.w;
  return f;
}

__global__ __launch_bounds__(512)
void mlp_fused_kernel(const float* __restrict__ x,  const float* __restrict__ W1,
                      const float* __restrict__ b1, const float* __restrict__ W2,
                      const float* __restrict__ b2, const float* __restrict__ W3,
                      const float* __restrict__ b3, float* __restrict__ out) {
  __shared__ __align__(16) _Float16 xs [M_TILE * XS_STRIDE];   // 9.2 KB
  __shared__ __align__(16) _Float16 h1s[M_TILE * HS_STRIDE];   // 33.8 KB
  __shared__ __align__(16) _Float16 h2s[M_TILE * HS_STRIDE];   // 33.8 KB
  __shared__ float ys[M_TILE * 16];                            // 4 KB

  const int tid  = threadIdx.x;
  const int lane = tid & 31;
  const int wv   = tid >> 5;                  // 0..15, one 16-col slice each
  const int ncol = (wv << 4) | (lane & 15);   // output column, layers 1&2
  const int koff = (lane & 16) ? 16 : 0;      // B-fragment per-lane K offset

  // ---- register-resident f16 B-fragments (vectorized f32 loads, once) ----
  v16h w1f[2], w2f[8], w3f;
  #pragma unroll
  for (int ks = 0; ks < 2; ++ks)
    w1f[ks] = load_bfrag(W1 + ncol * IN_DIM, ks * 32 + koff);
  #pragma unroll
  for (int ks = 0; ks < 8; ++ks)
    w2f[ks] = load_bfrag(W2 + ncol * H_DIM, ks * 32 + koff);
  const int ks3 = wv & 7;                     // this wave's K-chunk, layer 3
  const int n3  = lane & 15;
  if (n3 < OUT_DIM) {
    w3f = load_bfrag(W3 + n3 * H_DIM, ks3 * 32 + koff);
  } else {
    #pragma unroll
    for (int h = 0; h < 16; ++h) w3f[h] = (_Float16)0.0f;
  }
  const float bias1 = b1[ncol];
  const float bias2 = b2[ncol];

  // ---- software-pipelined x staging: tile t computes while t+grid loads ----
  const float4* __restrict__ xg = (const float4*)x;
  const int maxIdx = BATCH * (IN_DIM / 4) - 1;   // last valid float4 index
  int tile = blockIdx.x;
  float4 pre0, pre1;
  if (tile < NTILES) {
    const int base = tile * M_TILE * (IN_DIM / 4);
    pre0 = xg[min(base + tid,       maxIdx)];
    pre1 = xg[min(base + tid + 512, maxIdx)];
  }

  for (; tile < NTILES; tile += gridDim.x) {
    const int rowBase = tile * M_TILE;

    // ---- store staged x (f32->f16) + zero ys (protected by loop-end sync) --
    #pragma unroll
    for (int i = 0; i < 2; ++i) {
      const int flat4 = tid + i * 512;          // float4 index inside tile
      const int r = flat4 >> 4, c = (flat4 & 15) << 2;
      float4 v = (i == 0) ? pre0 : pre1;
      if (rowBase + r >= BATCH) v = make_float4(0.f, 0.f, 0.f, 0.f);
      h4 hv; hv[0]=(_Float16)v.x; hv[1]=(_Float16)v.y;
             hv[2]=(_Float16)v.z; hv[3]=(_Float16)v.w;
      *(h4*)&xs[r * XS_STRIDE + c] = hv;
    }
    ys[tid] = 0.f; ys[tid + 512] = 0.f;
    __syncthreads();

    // ---- prefetch next tile's x into registers (uniform branch) ----
    const int ntile = tile + gridDim.x;
    if (ntile < NTILES) {
      const int nb = ntile * M_TILE * (IN_DIM / 4);
      pre0 = xg[min(nb + tid,       maxIdx)];
      pre1 = xg[min(nb + tid + 512, maxIdx)];
    }

    // ---- layer 1: h1 = relu(x @ W1^T + b1) ----
    {
      v8f acc[N_MT];
      #pragma unroll
      for (int mt = 0; mt < N_MT; ++mt) acc[mt] = (v8f){};
      #pragma unroll
      for (int ks = 0; ks < 2; ++ks) {
        v16h a[N_MT];
        #pragma unroll
        for (int mt = 0; mt < N_MT; ++mt)
          a[mt] = make_afrag(&xs[mt * 16 * XS_STRIDE], XS_STRIDE, lane, ks);
        #pragma unroll
        for (int mt = 0; mt < N_MT; ++mt)
          acc[mt] = wmma_f16(a[mt], w1f[ks], acc[mt]);
      }
      #pragma unroll
      for (int mt = 0; mt < N_MT; ++mt)
        #pragma unroll
        for (int r = 0; r < 8; ++r) {
          const int row = mt * 16 + r + ((lane & 16) ? 8 : 0);
          const float v = acc[mt][r] + bias1;
          h1s[row * HS_STRIDE + ncol] = (_Float16)(v > 0.f ? v : 0.f);
        }
    }
    __syncthreads();

    // ---- layer 2: h2 = relu(h1 @ W2^T + b2) ----
    {
      v8f acc[N_MT];
      #pragma unroll
      for (int mt = 0; mt < N_MT; ++mt) acc[mt] = (v8f){};
      #pragma unroll
      for (int ks = 0; ks < 8; ++ks) {
        v16h a[N_MT];
        #pragma unroll
        for (int mt = 0; mt < N_MT; ++mt)
          a[mt] = make_afrag(&h1s[mt * 16 * HS_STRIDE], HS_STRIDE, lane, ks);
        #pragma unroll
        for (int mt = 0; mt < N_MT; ++mt)
          acc[mt] = wmma_f16(a[mt], w2f[ks], acc[mt]);
      }
      #pragma unroll
      for (int mt = 0; mt < N_MT; ++mt)
        #pragma unroll
        for (int r = 0; r < 8; ++r) {
          const int row = mt * 16 + r + ((lane & 16) ? 8 : 0);
          const float v = acc[mt][r] + bias2;
          h2s[row * HS_STRIDE + ncol] = (_Float16)(v > 0.f ? v : 0.f);
        }
    }
    __syncthreads();

    // ---- layer 3: 8 K-chunks x 4 M-subtiles = 32 jobs over 16 waves ----
    #pragma unroll
    for (int jj = 0; jj < 2; ++jj) {
      const int mt3 = (wv >> 3) + jj * 2;
      v16h a = make_afrag(&h2s[mt3 * 16 * HS_STRIDE], HS_STRIDE, lane, ks3);
      v8f acc = {};
      acc = wmma_f16(a, w3f, acc);
      #pragma unroll
      for (int r = 0; r < 8; ++r) {
        const int row = mt3 * 16 + r + ((lane & 16) ? 8 : 0);
        atomicAdd(&ys[row * 16 + n3], acc[r]);     // ds_add_f32
      }
    }
    __syncthreads();

    // ---- epilogue: +b3, relu, row-normalize to 100, coalesced store ----
    if (tid < M_TILE && rowBase + tid < BATCH) {
      const int row = tid;
      float v[8], s = 0.f;
      #pragma unroll
      for (int c = 0; c < 8; ++c) {
        float t = ys[row * 16 + c] + b3[c];
        t = t > 0.f ? t : 0.f;
        v[c] = t; s += t;
      }
      const float scale = 100.f / s;
      float4* op = (float4*)(out + (size_t)(rowBase + row) * OUT_DIM);
      op[0] = make_float4(v[0]*scale, v[1]*scale, v[2]*scale, v[3]*scale);
      op[1] = make_float4(v[4]*scale, v[5]*scale, v[6]*scale, v[7]*scale);
    }
    __syncthreads();
  }
}

extern "C" void kernel_launch(void* const* d_in, const int* in_sizes, int n_in,
                              void* d_out, int out_size, void* d_ws, size_t ws_size,
                              hipStream_t stream) {
  (void)in_sizes; (void)n_in; (void)out_size; (void)d_ws; (void)ws_size;
  const float* x  = (const float*)d_in[0];
  const float* W1 = (const float*)d_in[1];
  const float* b1 = (const float*)d_in[2];
  const float* W2 = (const float*)d_in[3];
  const float* b2 = (const float*)d_in[4];
  const float* W3 = (const float*)d_in[5];
  const float* b3 = (const float*)d_in[6];
  float* out = (float*)d_out;
  hipLaunchKernelGGL(mlp_fused_kernel, dim3(1024), dim3(512), 0, stream,
                     x, W1, b1, W2, b2, W3, b3, out);
}
